// RNNBlock_74689481277982
// MI455X (gfx1250) — matrix operand order repaired and checked by reading, not compiled, not run on hardware
//
#include <hip/hip_runtime.h>

// RNNBlock fused 2-layer SimpleRNN for MI455X (gfx1250, wave32, WMMA).
// B=16384, Z=79, F=8, C=256.  out[:,t,:] = state after consuming x[:,78..t,:].
// Each wave owns 16 batch rows; states in per-wave LDS (f16); weights pre-packed
// to f16 WMMA B-fragment layout in workspace (L2-resident).

typedef _Float16 v16h __attribute__((ext_vector_type(16)));
typedef _Float16 v8h  __attribute__((ext_vector_type(8)));
typedef float    v8f  __attribute__((ext_vector_type(8)));

#define Bsz 16384
#define Zt  79
#define Fin 8
#define Ch  256

union V16 { v16h v; v8h h[2]; };

static __device__ __forceinline__ v8f wmma16(v16h a, v16h b, v8f c) {
  // (neg_a, A, neg_b, B, c_mod, C, reuse_a, reuse_b)
  return __builtin_amdgcn_wmma_f32_16x16x32_f16(false, a, false, b, (short)0, c,
                                                false, false);
}

// Select-form relu: lowers to a single v_max_num_f32 (no NaN canonicalize pass).
static __device__ __forceinline__ float relu(float v) { return v > 0.f ? v : 0.f; }

// B-fragment: 32 lanes x 16 f16 contiguous per lane (32 B) -> 2x global_load_b128
static __device__ __forceinline__ v16h load_frag_g(const _Float16* __restrict__ p) {
  V16 u;
  u.h[0] = *reinterpret_cast<const v8h*>(p);
  u.h[1] = *reinterpret_cast<const v8h*>(p + 8);
  return u.v;
}

// ---------------------------------------------------------------------------
// Weight packer: W [K][256] f32 row-major -> f16 B-fragment buffer.
// frag index = ((nt*Ktiles + kc)*512) + lane*16 + j, where
//   k = kc*32 + (j>>3)*16 + (lane>>4)*8 + (j&7) ; n = nt*16 + (lane&15); value = W[k][n] (0 if k>=K)
__global__ void pack_w(const float* __restrict__ W, _Float16* __restrict__ frag,
                       int K, int Ktiles) {
  int idx = blockIdx.x * blockDim.x + threadIdx.x;
  int total = 16 * Ktiles * 512;
  if (idx >= total) return;
  int nt = idx / (Ktiles * 512);
  int r  = idx - nt * (Ktiles * 512);
  int kc = r / 512;
  int r2 = r - kc * 512;
  int ln = r2 >> 4;
  int j  = r2 & 15;
  int k  = kc * 32 + (j >> 3) * 16 + (ln >> 4) * 8 + (j & 7);
  int n  = nt * 16 + (ln & 15);
  frag[idx] = (k < K) ? (_Float16)W[(long)k * Ch + n] : (_Float16)0.f;
}

// ---------------------------------------------------------------------------
__global__ __launch_bounds__(64) void rnn_fused(
    const float* __restrict__ x, const float* __restrict__ b1,
    const float* __restrict__ b2,
    const _Float16* __restrict__ wx1f, const _Float16* __restrict__ wh1f,
    const _Float16* __restrict__ wx2f, const _Float16* __restrict__ wh2f,
    float* __restrict__ out) {
  __shared__ __attribute__((aligned(16))) _Float16 h1buf[2][16][Ch];
  __shared__ __attribute__((aligned(16))) _Float16 h2buf[2][16][Ch];
  __shared__ __attribute__((aligned(16))) _Float16 xstage[2][16][32];

  const int lane = threadIdx.x & 31;
  const int wv   = threadIdx.x >> 5;
  const int mrow = lane & 15;   // row within tile (lanes 0-15 / 16-31 mirror)
  const int hi   = lane >> 4;   // half-wave selector
  const long m0  = ((long)blockIdx.x * 2 + wv) * 16;  // first batch row of tile

  // zero initial states + zero-pad x staging (K 8..31 stay zero forever)
  {
    _Float16* p1 = &h1buf[wv][0][0];
    _Float16* p2 = &h2buf[wv][0][0];
    for (int i = lane; i < 16 * Ch; i += 32) { p1[i] = (_Float16)0.f; p2[i] = (_Float16)0.f; }
    _Float16* p3 = &xstage[wv][0][0];
    for (int i = lane; i < 16 * 32; i += 32) p3[i] = (_Float16)0.f;
  }

  const _Float16* px1 = wx1f + lane * 16;
  const _Float16* ph1 = wh1f + lane * 16;
  const _Float16* px2 = wx2f + lane * 16;
  const _Float16* ph2 = wh2f + lane * 16;

  for (int t = Zt - 1; t >= 0; --t) {
    // ---- stage x_t (16 rows x 8 f32 -> f16, K-padded to 32) ----
    if (hi == 0) {
      const float* xp = x + ((m0 + mrow) * Zt + t) * Fin;
      v8h xh;
#pragma unroll
      for (int f = 0; f < 8; ++f) xh[f] = (_Float16)xp[f];
      *reinterpret_cast<v8h*>(&xstage[wv][mrow][0]) = xh;
    }
    // A-fragment (16x32 f16): lanes0-15 hold K0..7|16..23, lanes16-31 K8..15|24..31
    V16 ax;
    ax.h[0] = *reinterpret_cast<const v8h*>(&xstage[wv][mrow][hi * 8]);
    ax.h[1] = *reinterpret_cast<const v8h*>(&xstage[wv][mrow][16 + hi * 8]);

    // h1_{t-1} A-fragments (8 K-chunks)
    V16 ah1[8];
#pragma unroll
    for (int kc = 0; kc < 8; ++kc) {
      const _Float16* r = &h1buf[wv][mrow][kc * 32];
      ah1[kc].h[0] = *reinterpret_cast<const v8h*>(r + hi * 8);
      ah1[kc].h[1] = *reinterpret_cast<const v8h*>(r + 16 + hi * 8);
    }

    // ---- layer 1: h1_t = relu(x_t@Wx1 + b1 + h1_{t-1}@Wh1) ----
    for (int nt = 0; nt < 16; ++nt) {
      float bias = b1[nt * 16 + mrow];
      v8f acc = {bias, bias, bias, bias, bias, bias, bias, bias};
      acc = wmma16(ax.v, load_frag_g(px1 + nt * 512), acc);
#pragma unroll
      for (int kc = 0; kc < 8; ++kc)
        acc = wmma16(ah1[kc].v, load_frag_g(ph1 + (nt * 8 + kc) * 512), acc);
#pragma unroll
      for (int i = 0; i < 8; ++i) {
        float v = relu(acc[i]);                       // C layout: m=hi*8+i, n=nt*16+mrow
        h1buf[wv][hi * 8 + i][nt * 16 + mrow] = (_Float16)v;
      }
    }

    // reload A-fragments: h1_t (just produced) and h2_{t-1}
    V16 ag[8], ah2[8];
#pragma unroll
    for (int kc = 0; kc < 8; ++kc) {
      const _Float16* r1 = &h1buf[wv][mrow][kc * 32];
      ag[kc].h[0] = *reinterpret_cast<const v8h*>(r1 + hi * 8);
      ag[kc].h[1] = *reinterpret_cast<const v8h*>(r1 + 16 + hi * 8);
      const _Float16* r2 = &h2buf[wv][mrow][kc * 32];
      ah2[kc].h[0] = *reinterpret_cast<const v8h*>(r2 + hi * 8);
      ah2[kc].h[1] = *reinterpret_cast<const v8h*>(r2 + 16 + hi * 8);
    }

    // ---- layer 2: h2_t = relu(h1_t@Wx2 + b2 + h2_{t-1}@Wh2) ; write out ----
    for (int nt = 0; nt < 16; ++nt) {
      float bias = b2[nt * 16 + mrow];
      v8f acc = {bias, bias, bias, bias, bias, bias, bias, bias};
#pragma unroll
      for (int kc = 0; kc < 8; ++kc)
        acc = wmma16(ag[kc].v, load_frag_g(px2 + (nt * 8 + kc) * 512), acc);
#pragma unroll
      for (int kc = 0; kc < 8; ++kc)
        acc = wmma16(ah2[kc].v, load_frag_g(ph2 + (nt * 8 + kc) * 512), acc);
#pragma unroll
      for (int i = 0; i < 8; ++i) {
        float v = relu(acc[i]);
        h2buf[wv][hi * 8 + i][nt * 16 + mrow] = (_Float16)v;
        out[((m0 + hi * 8 + i) * Zt + t) * (long)Ch + nt * 16 + mrow] = v;
      }
    }
  }
}

// ---------------------------------------------------------------------------
extern "C" void kernel_launch(void* const* d_in, const int* in_sizes, int n_in,
                              void* d_out, int out_size, void* d_ws, size_t ws_size,
                              hipStream_t stream) {
  const float* x   = (const float*)d_in[0];
  const float* Wx1 = (const float*)d_in[1];
  const float* Wh1 = (const float*)d_in[2];
  const float* b1  = (const float*)d_in[3];
  const float* Wx2 = (const float*)d_in[4];
  const float* Wh2 = (const float*)d_in[5];
  const float* b2  = (const float*)d_in[6];
  float* out = (float*)d_out;

  // workspace layout (f16 elements): wx1f[8192] wh1f[65536] wx2f[65536] wh2f[65536]
  _Float16* ws   = (_Float16*)d_ws;
  _Float16* wx1f = ws;
  _Float16* wh1f = ws + 8192;
  _Float16* wx2f = ws + 73728;
  _Float16* wh2f = ws + 139264;

  pack_w<<<(16 * 1 * 512 + 255) / 256, 256, 0, stream>>>(Wx1, wx1f, Fin, 1);
  pack_w<<<(16 * 8 * 512 + 255) / 256, 256, 0, stream>>>(Wh1, wh1f, Ch, 8);
  pack_w<<<(16 * 8 * 512 + 255) / 256, 256, 0, stream>>>(Wx2, wx2f, Ch, 8);
  pack_w<<<(16 * 8 * 512 + 255) / 256, 256, 0, stream>>>(Wh2, wh2f, Ch, 8);

  // 16384 rows / (16 rows per wave * 2 waves per block) = 512 blocks of 64 threads
  rnn_fused<<<512, 64, 0, stream>>>(x, b1, b2, wx1f, wh1f, wx2f, wh2f, out);
}
